// QRNNLayer_17738214932828
// MI455X (gfx1250) — matrix operand order repaired and checked by reading, not compiled, not run on hardware
//
#include <hip/hip_runtime.h>

// CDNA5 WMMA fragment types (probe-confirmed signatures)
typedef __bf16 v16bf __attribute__((ext_vector_type(16)));
typedef float  v8f   __attribute__((ext_vector_type(8)));

union FragBF {
    v16bf v;
    uint4 q[2];
    unsigned short u[16];
};

// ---------------------------------------------------------------------------
// Async global->LDS copy (CDNA5 GLOBAL_LOAD_ASYNC_TO_LDS_B128, ASYNCcnt).
// Builtin signature (from hipcc diagnostic): (v4i AS1* src, v4i AS3* dst,
// imm offset, imm cpol).  Guarded: falls back to sync b128 copy if absent.
// ---------------------------------------------------------------------------
#if defined(__has_builtin)
#  if __has_builtin(__builtin_amdgcn_global_load_async_to_lds_b128)
#    define USE_ASYNC_COPY 1
#  else
#    define USE_ASYNC_COPY 0
#  endif
#  if __has_builtin(__builtin_amdgcn_s_wait_asynccnt)
#    define HAVE_WAIT_ASYNC_BUILTIN 1
#  else
#    define HAVE_WAIT_ASYNC_BUILTIN 0
#  endif
#else
#  define USE_ASYNC_COPY 0
#  define HAVE_WAIT_ASYNC_BUILTIN 0
#endif

typedef int v4i_vec __attribute__((vector_size(16)));
typedef __attribute__((address_space(1))) v4i_vec* gas_v4i_ptr;   // global
typedef __attribute__((address_space(3))) v4i_vec* las_v4i_ptr;   // LDS

__device__ __forceinline__ void copy_b128_to_lds(const unsigned short* g, unsigned short* l) {
#if USE_ASYNC_COPY
    __builtin_amdgcn_global_load_async_to_lds_b128(
        (gas_v4i_ptr)g, (las_v4i_ptr)l, 0, 0);
#else
    *(uint4*)l = *(const uint4*)g;
#endif
}

__device__ __forceinline__ void wait_async_copies() {
#if USE_ASYNC_COPY
#  if HAVE_WAIT_ASYNC_BUILTIN
    __builtin_amdgcn_s_wait_asynccnt(0);
#  else
    asm volatile("s_wait_asynccnt 0" ::: "memory");
#  endif
#endif
}

// ---------------------------------------------------------------------------
// Helpers
// ---------------------------------------------------------------------------
__device__ __forceinline__ unsigned short f32_to_bf16(float f) {
    unsigned int u = __float_as_uint(f);
    unsigned int r = (u + 0x7FFFu + ((u >> 16) & 1u)) >> 16;  // round-nearest-even
    return (unsigned short)r;
}
__device__ __forceinline__ float fast_sigmoid(float x) {
    return 1.0f / (1.0f + __expf(-x));
}
__device__ __forceinline__ float fast_tanh(float x) {
    return 2.0f / (1.0f + __expf(-2.0f * x)) - 1.0f;
}

// ---------------------------------------------------------------------------
// Kernel 1: shift-transpose-cast  x[b][c][t] (f32) -> xT[b][t][q'] (bf16)
//   q' = k*256 + c  (tap-blocked):  xT[b][t][c]     = x[b][c][t-1]   (k=0)
//                                   xT[b][t][256+c] = x[b][c][t]     (k=1)
// ---------------------------------------------------------------------------
#define TLD 74   // ushort row stride for the 65-column tile (bank-conflict-free)

__global__ __launch_bounds__(256) void shift_transpose_kernel(
    const float* __restrict__ x, unsigned short* __restrict__ xT)
{
    __shared__ unsigned short tile[64 * TLD];   // tile[c][j], j=0 -> t0-1
    const int tt0 = blockIdx.x * 64;
    const int c0  = blockIdx.y * 64;
    const int b   = blockIdx.z;
    const int tid = threadIdx.x;
    const float* xr = x + (b * 256 + c0) * 4096;

    // phase 1: main 64 columns, coalesced float4 reads
    {
        const int r  = tid >> 2;           // c-local 0..63
        const int jc = (tid & 3) * 16;     // t chunk
        const float* row = xr + r * 4096 + tt0 + jc;
        #pragma unroll
        for (int i = 0; i < 4; ++i) {
            float4 f = *(const float4*)(row + i * 4);
            const int base = r * TLD + 1 + jc + i * 4;
            tile[base + 0] = f32_to_bf16(f.x);
            tile[base + 1] = f32_to_bf16(f.y);
            tile[base + 2] = f32_to_bf16(f.z);
            tile[base + 3] = f32_to_bf16(f.w);
        }
    }
    // phase 1b: boundary column t0-1 (zero at t=-1)
    if (tid < 64) {
        float v = (tt0 > 0) ? xr[tid * 4096 + tt0 - 1] : 0.f;
        tile[tid * TLD] = f32_to_bf16(v);
    }
    __syncthreads();

    // phase 2: write both tap blocks, 2x2 uint4 coalesced stores per thread
    {
        const int t  = tid >> 2;           // 0..63
        const int cg = (tid & 3) * 16;     // c chunk of 16
        union { uint4 q[2]; unsigned short u[16]; } o0, o1;
        #pragma unroll
        for (int i = 0; i < 16; ++i) {
            const int cc = cg + i;
            o0.u[i] = tile[cc * TLD + t];       // k=0: x[c][t-1]
            o1.u[i] = tile[cc * TLD + t + 1];   // k=1: x[c][t]
        }
        unsigned short* dst = xT + (b * 4096 + tt0 + t) * 512;
        *(uint4*)(dst + c0 + cg)           = o0.q[0];
        *(uint4*)(dst + c0 + cg + 8)       = o0.q[1];
        *(uint4*)(dst + 256 + c0 + cg)     = o1.q[0];
        *(uint4*)(dst + 256 + c0 + cg + 8) = o1.q[1];
    }
}

// ---------------------------------------------------------------------------
// Kernel 2: repack W[h][c][k] (f32) -> Wq[g][h][q'=k*256+c] (bf16)
// ---------------------------------------------------------------------------
__global__ __launch_bounds__(256) void repack_w_kernel(
    const float* __restrict__ Wz, const float* __restrict__ Wf,
    const float* __restrict__ Wo, unsigned short* __restrict__ wq)
{
    const int i = blockIdx.x * 256 + threadIdx.x;      // 0 .. 3*131072-1
    if (i < 3 * 131072) {
        const int g = i >> 17;
        const int r = i & 131071;
        const int h = r >> 9;
        const int q = r & 511;
        const int k = q >> 8;
        const int c = q & 255;
        const float* W = (g == 0) ? Wz : ((g == 1) ? Wf : Wo);
        wq[i] = f32_to_bf16(W[h * 512 + c * 2 + k]);
    }
}

// ---------------------------------------------------------------------------
// Kernel 3: gate GEMM via v_wmma_f32_16x16x32_bf16
//   D[g][b][h][t] = act( A[h][q'] x B[q'][t] + bias ),  B[q'][t] = xT[b][t][q']
// Block tile 64(M=h) x 128(N=t).  MI455X has 320KB LDS/WGP, so the FULL
// A (64x512) and B (128x512) bf16 tiles are loaded once per block with
// async global->LDS b128 copies (one s_wait_asynccnt + one barrier), then
// all 16 K-slabs x 4 WMMAs per wave run back-to-back from LDS.
// Row stride 520 ushorts (260 words ~ 4 mod 64) keeps b128 frag loads
// on distinct banks. ~195KB dynamic LDS -> 1 block (8 waves) per WGP.
// ---------------------------------------------------------------------------
#define GLD 520                         // LDS row stride in ushorts (16B aligned)
#define LDS_A_USHORTS (64 * GLD)        // 33,280
#define LDS_B_USHORTS (128 * GLD)       // 66,560
#define GEMM_LDS_BYTES ((LDS_A_USHORTS + LDS_B_USHORTS) * 2)   // 199,680 B

__global__ __launch_bounds__(256) void qrnn_gate_gemm_kernel(
    const unsigned short* __restrict__ xT,   // [8][4096][512] bf16
    const unsigned short* __restrict__ wq,   // [3][256][512]  bf16
    const float* __restrict__ bz,
    const float* __restrict__ bfg,
    const float* __restrict__ bo,
    float* __restrict__ gates)               // [3][8][256][4096] f32
{
    extern __shared__ unsigned short smem[];
    unsigned short* lA = smem;                    // [m=64][q=512] stride GLD
    unsigned short* lB = smem + LDS_A_USHORTS;    // [n=128][q=512] stride GLD

    const int t0  = blockIdx.x * 128;
    const int h0  = blockIdx.y * 64;
    const int g   = blockIdx.z % 3;              // 0=z, 1=f, 2=o
    const int b   = blockIdx.z / 3;

    const int tid  = threadIdx.x;
    const int lane = tid & 31;
    const int w    = tid >> 5;
    const int m0   = (w & 3) * 16;
    const int n0   = (w >> 2) * 64;
    const int hi   = lane >> 4;
    const int l16  = lane & 15;

    // ---- stage full A tile: 64 rows x 512, 16 async b128 per thread ----
    {
        const int r  = tid >> 2;                 // 0..63
        const int cg = (tid & 3) * 128;          // column group
        const unsigned short* src = wq + g * (256 * 512) + (h0 + r) * 512 + cg;
        unsigned short* dst = lA + r * GLD + cg;
        #pragma unroll
        for (int i = 0; i < 16; ++i)
            copy_b128_to_lds(src + i * 8, dst + i * 8);
    }
    // ---- stage full B tile: 128 rows x 512, 32 async b128 per thread ----
    {
        const int r  = tid >> 1;                 // 0..127
        const int cg = (tid & 1) * 256;
        const unsigned short* src = xT + (b * 4096 + t0 + r) * 512 + cg;
        unsigned short* dst = lB + r * GLD + cg;
        #pragma unroll
        for (int i = 0; i < 32; ++i)
            copy_b128_to_lds(src + i * 8, dst + i * 8);
    }
    wait_async_copies();
    __syncthreads();

    const v8f vzero = {0.f, 0.f, 0.f, 0.f, 0.f, 0.f, 0.f, 0.f};
    v8f acc[4];
    acc[0] = vzero; acc[1] = vzero; acc[2] = vzero; acc[3] = vzero;

    const int arow = (m0 + l16) * GLD;
    #pragma unroll
    for (int kk = 0; kk < 16; ++kk) {            // 16 K-slabs of 32 over q'=512
        const int q0 = kk * 32;
        FragBF fa;
        fa.q[0] = *(const uint4*)&lA[arow + q0 + 8 * hi];
        fa.q[1] = *(const uint4*)&lA[arow + q0 + 16 + 8 * hi];
        #pragma unroll
        for (int s = 0; s < 4; ++s) {
            FragBF fb;
            const int brow = (n0 + s * 16 + l16) * GLD + q0 + 8 * hi;
            fb.q[0] = *(const uint4*)&lB[brow];
            fb.q[1] = *(const uint4*)&lB[brow + 16];
            acc[s] = __builtin_amdgcn_wmma_f32_16x16x32_bf16(
                         false, fa.v, false, fb.v, (short)0, acc[s], false, false);
        }
    }

    // epilogue: bias + activation fused here so the scan is a pure fma chain
    const float* bias = (g == 0) ? bz : ((g == 1) ? bfg : bo);
    #pragma unroll
    for (int v = 0; v < 8; ++v) {
        const int gh = h0 + m0 + v + 8 * hi;
        const float bb = bias[gh];
        const int rowbase = ((g * 8 + b) * 256 + gh) * 4096 + t0 + n0;
        #pragma unroll
        for (int s = 0; s < 4; ++s) {
            float xv = acc[s][v] + bb;
            xv = (g == 0) ? fast_tanh(xv) : fast_sigmoid(xv);
            gates[rowbase + s * 16 + l16] = xv;
        }
    }
}

// ---------------------------------------------------------------------------
// Scan: chunked parallel linear recurrence  c_t = f*(c_{t-1}-z) + z
// 16 chunks of 256 steps; serial depth 256+16 instead of 4096.
// ---------------------------------------------------------------------------
__global__ __launch_bounds__(256) void scan_pass1_kernel(
    const float* __restrict__ gates, float* __restrict__ cA, float* __restrict__ cB)
{
    const int chunk = blockIdx.x & 15;
    const int b     = blockIdx.x >> 4;
    const int h     = threadIdx.x;
    const int chain = b * 256 + h;
    const float4* zr = (const float4*)(gates + ((0 * 8 + b) * 256 + h) * 4096) + chunk * 64;
    const float4* fr = (const float4*)(gates + ((1 * 8 + b) * 256 + h) * 4096) + chunk * 64;
    float c = 0.f, A = 1.f;
    for (int i = 0; i < 64; ++i) {
        __builtin_prefetch(&zr[i + 8], 0, 0);
        __builtin_prefetch(&fr[i + 8], 0, 0);
        const float4 z = zr[i], f = fr[i];
        c = f.x * (c - z.x) + z.x;  A *= f.x;
        c = f.y * (c - z.y) + z.y;  A *= f.y;
        c = f.z * (c - z.z) + z.z;  A *= f.z;
        c = f.w * (c - z.w) + z.w;  A *= f.w;
    }
    cA[chain * 16 + chunk] = A;
    cB[chain * 16 + chunk] = c;
}

__global__ __launch_bounds__(256) void scan_pass2_kernel(
    const float* __restrict__ cA, const float* __restrict__ cB,
    float* __restrict__ centry)
{
    const int chain = blockIdx.x * 256 + threadIdx.x;   // 2048 chains
    float c = 0.f;
    #pragma unroll
    for (int j = 0; j < 16; ++j) {
        centry[chain * 16 + j] = c;
        c = cA[chain * 16 + j] * c + cB[chain * 16 + j];
    }
}

__global__ __launch_bounds__(256) void scan_pass3_kernel(
    const float* __restrict__ gates, const float* __restrict__ centry,
    float* __restrict__ out)
{
    const int chunk = blockIdx.x & 15;
    const int b     = blockIdx.x >> 4;
    const int h     = threadIdx.x;
    const int chain = b * 256 + h;
    const float4* zr = (const float4*)(gates + ((0 * 8 + b) * 256 + h) * 4096) + chunk * 64;
    const float4* fr = (const float4*)(gates + ((1 * 8 + b) * 256 + h) * 4096) + chunk * 64;
    const float4* og = (const float4*)(gates + ((2 * 8 + b) * 256 + h) * 4096) + chunk * 64;
    float4* hr = (float4*)(out + chain * 4096) + chunk * 64;
    float4* cr = (float4*)(out + 8 * 256 * 4096 + chain * 4096) + chunk * 64;

    float c = centry[chain * 16 + chunk];
    for (int i = 0; i < 64; ++i) {
        __builtin_prefetch(&zr[i + 8], 0, 0);
        __builtin_prefetch(&fr[i + 8], 0, 0);
        __builtin_prefetch(&og[i + 8], 0, 0);
        const float4 z = zr[i], f = fr[i], o = og[i];
        float4 hv, cv;
        c = f.x * (c - z.x) + z.x; cv.x = c; hv.x = o.x * c;
        c = f.y * (c - z.y) + z.y; cv.y = c; hv.y = o.y * c;
        c = f.z * (c - z.z) + z.z; cv.z = c; hv.z = o.z * c;
        c = f.w * (c - z.w) + z.w; cv.w = c; hv.w = o.w * c;
        hr[i] = hv;
        cr[i] = cv;
    }
}

// ---------------------------------------------------------------------------
// Launch
// ---------------------------------------------------------------------------
extern "C" void kernel_launch(void* const* d_in, const int* in_sizes, int n_in,
                              void* d_out, int out_size, void* d_ws, size_t ws_size,
                              hipStream_t stream) {
    const float* x   = (const float*)d_in[0];
    const float* Wz  = (const float*)d_in[1];
    const float* bz  = (const float*)d_in[2];
    const float* Wf  = (const float*)d_in[3];
    const float* bfg = (const float*)d_in[4];
    const float* Wo  = (const float*)d_in[5];
    const float* bo  = (const float*)d_in[6];
    // d_in[7], d_in[8] (Wi, bi) unused: reference deletes i.

    // workspace layout
    float*          gates  = (float*)d_ws;                   // 25,165,824 f32 (96 MiB)
    unsigned short* xT     = (unsigned short*)(gates + 25165824);  // 16,777,216 bf16 (32 MiB)
    unsigned short* wqb    = xT + 16777216;                  // 393,216 bf16
    float*          cA     = (float*)(wqb + 393216);         // 32,768 f32
    float*          cB     = cA + 32768;                     // 32,768 f32
    float*          centry = cB + 32768;                     // 32,768 f32

    // allow >64KB dynamic LDS for the GEMM kernel (320KB/WGP on CDNA5)
    (void)hipFuncSetAttribute((const void*)qrnn_gate_gemm_kernel,
                              hipFuncAttributeMaxDynamicSharedMemorySize,
                              GEMM_LDS_BYTES);

    // 1) shift-transpose-cast x -> xT[b][t][q']
    dim3 tr_grid(4096 / 64, 256 / 64, 8);
    shift_transpose_kernel<<<tr_grid, 256, 0, stream>>>(x, xT);

    // 2) repack weights -> Wq[g][h][q']
    repack_w_kernel<<<(3 * 131072) / 256, 256, 0, stream>>>(Wz, Wf, Wo, wqb);

    // 3) WMMA gate GEMM (+bias+activation), whole-tile async-LDS staging
    dim3 gemm_grid(4096 / 128, 256 / 64, 3 * 8);
    qrnn_gate_gemm_kernel<<<gemm_grid, 256, GEMM_LDS_BYTES, stream>>>(
        xT, wqb, bz, bfg, bo, gates);

    // 4) chunked fo-pooling scan
    scan_pass1_kernel<<<128, 256, 0, stream>>>(gates, cA, cB);
    scan_pass2_kernel<<<8, 256, 0, stream>>>(cA, cB, centry);
    scan_pass3_kernel<<<128, 256, 0, stream>>>(gates, centry, (float*)d_out);
}
